// AdvancedTransformerBlock_71923522339239
// MI455X (gfx1250) — compile-verified, matrix operand order checked by reading, not run on hardware
//
#include <hip/hip_runtime.h>
#include <hip/hip_bf16.h>
#include <math.h>

// ---------------------------------------------------------------------------
// AdvancedTransformerBlock for MI455X (gfx1250, wave32, WMMA).
// All matmuls use v_wmma_f32_16x16x32_bf16 (bf16 inputs, f32 accum).
// Tile staging uses GLOBAL_LOAD_ASYNC_TO_LDS (ASYNCcnt) + double buffering.
// ---------------------------------------------------------------------------

typedef __attribute__((ext_vector_type(16))) __bf16 v16bf;
typedef __attribute__((ext_vector_type(8)))  float  v8f;

#define S_LEN   2048
#define BATCH   2
#define DMODEL  1024
#define NHEADS  16
#define NKV     4
#define HEADDIM 64
#define NROWS   (S_LEN * BATCH)   // 4096
#define DFF     (4 * DMODEL)      // 4096
#define EPSV    1e-5f
#define QK_SCALE 0.03125f         // D ** -0.5
#define NEG_INF (-3.0e38f)

#if __has_builtin(__builtin_amdgcn_global_load_async_to_lds_b128)
#define HAVE_ASYNC_LDS 1
#else
#define HAVE_ASYNC_LDS 0
#endif

// ---- bf16 <-> f32 via explicit bit ops (round-to-nearest-even) ------------
__device__ __forceinline__ __bf16 f2bf(float f) {
    unsigned u = __builtin_bit_cast(unsigned, f);
    unsigned r = (u + 0x7FFFu + ((u >> 16) & 1u)) >> 16;
    unsigned short s = (unsigned short)r;
    return __builtin_bit_cast(__bf16, s);
}

// A-fragment (16-bit, 16x32) K index for vector element e, lane-half hi.
__device__ __forceinline__ int akidx(int e, int hi) {
    return (e < 8 ? e : e + 8) + 8 * hi;
}

__device__ __forceinline__ v8f v8f_zero() {
    v8f z;
    #pragma unroll
    for (int i = 0; i < 8; ++i) z[i] = 0.0f;
    return z;
}

// ---- async global -> LDS 16B copy (per lane) ------------------------------
// Builtin signature (from compiler diagnostic): param pointee type is the
// GCC vector 'int __attribute__((vector_size(16)))', global AS src, LDS dst.
typedef int v4i_vs __attribute__((vector_size(16)));
typedef __attribute__((address_space(1))) v4i_vs* g_v4i_ptr;
typedef __attribute__((address_space(3))) v4i_vs* l_v4i_ptr;

__device__ __forceinline__ void async_cp16(const __bf16* g, __bf16* l) {
#if HAVE_ASYNC_LDS
    __builtin_amdgcn_global_load_async_to_lds_b128(
        (g_v4i_ptr)(const_cast<__bf16*>(g)), (l_v4i_ptr)(l), 0, 0);
#else
    *reinterpret_cast<uint4*>(l) = *reinterpret_cast<const uint4*>(g);
#endif
}

__device__ __forceinline__ void wait_async() {
#if HAVE_ASYNC_LDS
 #if __has_builtin(__builtin_amdgcn_s_wait_asynccnt)
    __builtin_amdgcn_s_wait_asynccnt(0);
 #else
    asm volatile("s_wait_asynccnt 0x0" ::: "memory");
 #endif
#endif
}

// ---------------------------------------------------------------------------
// f32 -> bf16 elementwise (weight conversion)
// ---------------------------------------------------------------------------
__global__ __launch_bounds__(256)
void f32_to_bf16_kernel(const float* __restrict__ in, __bf16* __restrict__ out, int n) {
    int i = blockIdx.x * 256 + threadIdx.x;
    if (i < n) out[i] = f2bf(in[i]);
}

// ---------------------------------------------------------------------------
// Row RMSNorm: f32 [NROWS, DMODEL] -> bf16, one block per row.
// ---------------------------------------------------------------------------
__global__ __launch_bounds__(256)
void rmsnorm_bf16_kernel(const float* __restrict__ x, const float* __restrict__ w,
                         __bf16* __restrict__ out) {
    __shared__ float red[8];
    const int r = blockIdx.x;
    const float* xr = x + (size_t)r * DMODEL;
    float vals[4];
    float ss = 0.0f;
    #pragma unroll
    for (int i = 0; i < 4; ++i) {
        vals[i] = xr[threadIdx.x + i * 256];
        ss += vals[i] * vals[i];
    }
    #pragma unroll
    for (int m = 1; m < 32; m <<= 1) ss += __shfl_xor(ss, m, 32);
    if ((threadIdx.x & 31) == 0) red[threadIdx.x >> 5] = ss;
    __syncthreads();
    float tot = 0.0f;
    #pragma unroll
    for (int i = 0; i < 8; ++i) tot += red[i];
    const float rms = rsqrtf(tot * (1.0f / DMODEL) + EPSV);
    #pragma unroll
    for (int i = 0; i < 4; ++i) {
        const int c = threadIdx.x + i * 256;
        out[(size_t)r * DMODEL + c] = f2bf(vals[i] * rms * w[c]);
    }
}

// ---------------------------------------------------------------------------
// WMMA GEMM: C[M,N] = A[M,K](bf16) * B[K,N](bf16) + bias, double-buffered
// async LDS staging. 256 thr = 8 waves; block tile 128x128, BK=32.
// ---------------------------------------------------------------------------
#define EPI_F32       0
#define EPI_F32_RES   1
#define EPI_BF16_GELU 2

template <int EPI>
__global__ __launch_bounds__(256)
void gemm_bf16_wmma(const __bf16* __restrict__ A, const __bf16* __restrict__ Bm,
                    const float* __restrict__ bias, const float* __restrict__ resid,
                    void* __restrict__ Cout, int M, int N, int K) {
    __shared__ alignas(16) __bf16 As[2][128 * 32];
    __shared__ alignas(16) __bf16 Bs[2][32 * 128];

    const int tid  = threadIdx.x;
    const int lane = tid & 31;
    const int wave = tid >> 5;
    const int hi   = lane >> 4;
    const int ln   = lane & 15;
    const int wm   = wave & 3;   // 4 waves along M (32 rows each)
    const int wn   = wave >> 2;  // 2 waves along N (64 cols each)
    const int m0   = blockIdx.x * 128;
    const int n0   = blockIdx.y * 128;

    // per-thread copy geometry: 2 x 16B chunks for each of A,B per tile
    const int c0  = tid * 2;
    const int ar0 = c0 >> 2,        ac0 = (c0 & 3) * 8;
    const int ar1 = (c0 + 1) >> 2,  ac1 = ((c0 + 1) & 3) * 8;
    const int br0 = c0 >> 4,        bc0 = (c0 & 15) * 8;
    const int br1 = (c0 + 1) >> 4,  bc1 = ((c0 + 1) & 15) * 8;

    v8f acc[2][4];
    #pragma unroll
    for (int i = 0; i < 2; ++i)
        #pragma unroll
        for (int j = 0; j < 4; ++j) acc[i][j] = v8f_zero();

    const int nk = K >> 5;

    // prefetch tile 0 into buffer 0
    {
        const int k0 = 0;
        async_cp16(&A[(size_t)(m0 + ar0) * K + k0 + ac0], &As[0][ar0 * 32 + ac0]);
        async_cp16(&A[(size_t)(m0 + ar1) * K + k0 + ac1], &As[0][ar1 * 32 + ac1]);
        async_cp16(&Bm[(size_t)(k0 + br0) * N + n0 + bc0], &Bs[0][br0 * 128 + bc0]);
        async_cp16(&Bm[(size_t)(k0 + br1) * N + n0 + bc1], &Bs[0][br1 * 128 + bc1]);
    }
    wait_async();
    __syncthreads();

    for (int t = 0; t < nk; ++t) {
        const int cur = t & 1;
        if (t + 1 < nk) {
            const int k0 = (t + 1) << 5;
            const int nxt = 1 - cur;
            async_cp16(&A[(size_t)(m0 + ar0) * K + k0 + ac0], &As[nxt][ar0 * 32 + ac0]);
            async_cp16(&A[(size_t)(m0 + ar1) * K + k0 + ac1], &As[nxt][ar1 * 32 + ac1]);
            async_cp16(&Bm[(size_t)(k0 + br0) * N + n0 + bc0], &Bs[nxt][br0 * 128 + bc0]);
            async_cp16(&Bm[(size_t)(k0 + br1) * N + n0 + bc1], &Bs[nxt][br1 * 128 + bc1]);
        }

        v16bf af[2], bfr[4];
        #pragma unroll
        for (int tm = 0; tm < 2; ++tm)
            #pragma unroll
            for (int e = 0; e < 16; ++e)
                af[tm][e] = As[cur][(wm * 32 + tm * 16 + ln) * 32 + akidx(e, hi)];
        #pragma unroll
        for (int tn = 0; tn < 4; ++tn)
            #pragma unroll
            for (int e = 0; e < 16; ++e)
                bfr[tn][e] = Bs[cur][akidx(e, hi) * 128 + wn * 64 + tn * 16 + ln];

        #pragma unroll
        for (int tm = 0; tm < 2; ++tm)
            #pragma unroll
            for (int tn = 0; tn < 4; ++tn)
                acc[tm][tn] = __builtin_amdgcn_wmma_f32_16x16x32_bf16(
                    false, af[tm], false, bfr[tn], (short)0, acc[tm][tn], false, false);

        wait_async();
        __syncthreads();
    }

    #pragma unroll
    for (int tm = 0; tm < 2; ++tm)
        #pragma unroll
        for (int tn = 0; tn < 4; ++tn)
            #pragma unroll
            for (int j = 0; j < 8; ++j) {
                const int row = m0 + wm * 32 + tm * 16 + j + 8 * hi;
                const int col = n0 + wn * 64 + tn * 16 + ln;
                const size_t idx = (size_t)row * N + col;
                float v = acc[tm][tn][j] + bias[col];
                if (EPI == EPI_F32) {
                    ((float*)Cout)[idx] = v;
                } else if (EPI == EPI_F32_RES) {
                    ((float*)Cout)[idx] = v + resid[idx];
                } else {
                    const float g = 0.5f * v * (1.0f + erff(v * 0.70710678118654752f));
                    ((__bf16*)Cout)[idx] = f2bf(g);
                }
            }
}

// ---------------------------------------------------------------------------
// QK per-head RMSNorm + RoPE + GQA relayout; V relayout.
// One 32-lane wave per (row, head); lane owns elements e and e+32 (RoPE pair).
// ---------------------------------------------------------------------------
__global__ __launch_bounds__(256)
void qkv_post_kernel(const float* __restrict__ qraw, const float* __restrict__ kraw,
                     const float* __restrict__ vraw, const float* __restrict__ qnw,
                     const float* __restrict__ knw,
                     __bf16* __restrict__ qb, __bf16* __restrict__ kb,
                     __bf16* __restrict__ vb) {
    const int unit = blockIdx.x * 8 + (threadIdx.x >> 5);
    const int lane = threadIdx.x & 31;
    const int NQ = NROWS * NHEADS;  // 65536
    const int NK = NROWS * NKV;     // 16384

    if (unit < NQ) {
        const int r = unit >> 4, h = unit & 15;
        const int s = r >> 1, b = r & 1;
        const float x1 = qraw[(size_t)r * DMODEL + h * 64 + lane];
        const float x2 = qraw[(size_t)r * DMODEL + h * 64 + lane + 32];
        float ss = x1 * x1 + x2 * x2;
        #pragma unroll
        for (int m = 1; m < 32; m <<= 1) ss += __shfl_xor(ss, m, 32);
        const float rms = rsqrtf(ss * (1.0f / HEADDIM) + EPSV);
        const float y1 = x1 * rms * qnw[lane];
        const float y2 = x2 * rms * qnw[lane + 32];
        const float f   = powf(10000.0f, -(float)lane / 32.0f);
        const float ang = (float)s * f;
        const float c = cosf(ang), sn = sinf(ang);
        const size_t base = (((size_t)(b * NHEADS + h)) * S_LEN + s) * 64;
        qb[base + lane]      = f2bf(y1 * c - y2 * sn);
        qb[base + lane + 32] = f2bf(y2 * c + y1 * sn);
    } else if (unit < NQ + NK) {
        const int u = unit - NQ;
        const int r = u >> 2, h = u & 3;
        const int s = r >> 1, b = r & 1;
        const float x1 = kraw[(size_t)r * (NKV * HEADDIM) + h * 64 + lane];
        const float x2 = kraw[(size_t)r * (NKV * HEADDIM) + h * 64 + lane + 32];
        float ss = x1 * x1 + x2 * x2;
        #pragma unroll
        for (int m = 1; m < 32; m <<= 1) ss += __shfl_xor(ss, m, 32);
        const float rms = rsqrtf(ss * (1.0f / HEADDIM) + EPSV);
        const float y1 = x1 * rms * knw[lane];
        const float y2 = x2 * rms * knw[lane + 32];
        const float f   = powf(10000.0f, -(float)lane / 32.0f);
        const float ang = (float)s * f;
        const float c = cosf(ang), sn = sinf(ang);
        const size_t base = (((size_t)(b * NKV + h)) * S_LEN + s) * 64;
        kb[base + lane]      = f2bf(y1 * c - y2 * sn);
        kb[base + lane + 32] = f2bf(y2 * c + y1 * sn);
    } else if (unit < NQ + 2 * NK) {
        const int u = unit - NQ - NK;
        const int r = u >> 2, h = u & 3;
        const int s = r >> 1, b = r & 1;
        const size_t base = (((size_t)(b * NKV + h)) * S_LEN + s) * 64;
        vb[base + lane]      = f2bf(vraw[(size_t)r * (NKV * HEADDIM) + h * 64 + lane]);
        vb[base + lane + 32] = f2bf(vraw[(size_t)r * (NKV * HEADDIM) + h * 64 + lane + 32]);
    }
}

// ---------------------------------------------------------------------------
// Flash attention (causal, GQA). Block = 128 thr (4 waves), 64 queries/block.
// QK^T and P*V via WMMA; online softmax in f32; async double-buffered K/V.
// ---------------------------------------------------------------------------
__global__ __launch_bounds__(128)
void attn_wmma_kernel(const __bf16* __restrict__ Q, const __bf16* __restrict__ Kc,
                      const __bf16* __restrict__ V, __bf16* __restrict__ Out) {
    __shared__ alignas(16) __bf16 Qs[64 * 64];
    __shared__ alignas(16) __bf16 Ks[2][64 * 64];
    __shared__ alignas(16) __bf16 Vs[2][64 * 64];
    __shared__ alignas(16) __bf16 Ps[4 * 16 * 64];

    const int tid  = threadIdx.x;
    const int lane = tid & 31;
    const int wave = tid >> 5;
    const int hi   = lane >> 4;
    const int ln   = lane & 15;
    const int q0   = blockIdx.x * 64;
    const int bh   = blockIdx.y;        // b*H + h
    const int b    = bh >> 4, h = bh & 15;
    const int kvh  = h >> 2;            // repeat_interleave GQA map

    const __bf16* qbase = Q + (((size_t)bh) * S_LEN + q0) * 64;
    const __bf16* kbase = Kc + (((size_t)(b * NKV + kvh)) * S_LEN) * 64;
    const __bf16* vbase = V  + (((size_t)(b * NKV + kvh)) * S_LEN) * 64;

    // async copies: 512 x 16B chunks per 64x64 tile over 128 threads
    #pragma unroll
    for (int i = 0; i < 4; ++i) {
        const int c = tid + i * 128;
        async_cp16(qbase + c * 8, Qs + c * 8);
        async_cp16(kbase + c * 8, &Ks[0][c * 8]);
        async_cp16(vbase + c * 8, &Vs[0][c * 8]);
    }
    wait_async();
    __syncthreads();

    v16bf aq[2];
    #pragma unroll
    for (int half = 0; half < 2; ++half)
        #pragma unroll
        for (int e = 0; e < 16; ++e)
            aq[half][e] = Qs[(wave * 16 + ln) * 64 + akidx(e, hi) + 32 * half];

    float mi[8], li[8];
    v8f o[4];
    #pragma unroll
    for (int j = 0; j < 8; ++j) { mi[j] = NEG_INF; li[j] = 0.0f; }
    #pragma unroll
    for (int t = 0; t < 4; ++t) o[t] = v8f_zero();

    const int nc = q0 / 64 + 1;   // causal: key chunks 0 .. diag
    for (int c = 0; c < nc; ++c) {
        const int cur = c & 1;
        if (c + 1 < nc) {
            const int nxt = 1 - cur;
            #pragma unroll
            for (int i = 0; i < 4; ++i) {
                const int ch = tid + i * 128;
                async_cp16(kbase + (size_t)(c + 1) * 4096 + ch * 8, &Ks[nxt][ch * 8]);
                async_cp16(vbase + (size_t)(c + 1) * 4096 + ch * 8, &Vs[nxt][ch * 8]);
            }
        }

        // scores: 16 queries x 64 keys per wave, K-dim = HEADDIM = 64
        v8f sc[4];
        #pragma unroll
        for (int tn = 0; tn < 4; ++tn) {
            sc[tn] = v8f_zero();
            v16bf bk0, bk1;
            #pragma unroll
            for (int e = 0; e < 16; ++e) {
                bk0[e] = Ks[cur][(tn * 16 + ln) * 64 + akidx(e, hi)];
                bk1[e] = Ks[cur][(tn * 16 + ln) * 64 + akidx(e, hi) + 32];
            }
            sc[tn] = __builtin_amdgcn_wmma_f32_16x16x32_bf16(
                false, aq[0], false, bk0, (short)0, sc[tn], false, false);
            sc[tn] = __builtin_amdgcn_wmma_f32_16x16x32_bf16(
                false, aq[1], false, bk1, (short)0, sc[tn], false, false);
        }

        // scale + causal mask + row max (rows spread across 16-lane halves)
        const bool diag = (c == nc - 1);
        float mcur[8];
        #pragma unroll
        for (int j = 0; j < 8; ++j) mcur[j] = NEG_INF;
        #pragma unroll
        for (int tn = 0; tn < 4; ++tn)
            #pragma unroll
            for (int j = 0; j < 8; ++j) {
                float v = sc[tn][j] * QK_SCALE;
                if (diag) {
                    const int row = q0 + wave * 16 + j + 8 * hi;
                    const int col = c * 64 + tn * 16 + ln;
                    if (col > row) v = NEG_INF;
                }
                sc[tn][j] = v;
                mcur[j] = fmaxf(mcur[j], v);
            }
        #pragma unroll
        for (int j = 0; j < 8; ++j)
            #pragma unroll
            for (int m = 1; m < 16; m <<= 1)
                mcur[j] = fmaxf(mcur[j], __shfl_xor(mcur[j], m, 32));

        float al[8], rs[8];
        #pragma unroll
        for (int j = 0; j < 8; ++j) {
            const float mn = fmaxf(mi[j], mcur[j]);
            al[j] = expf(mi[j] - mn);
            mi[j] = mn;
            rs[j] = 0.0f;
        }
        #pragma unroll
        for (int tn = 0; tn < 4; ++tn)
            #pragma unroll
            for (int j = 0; j < 8; ++j) {
                const float p = expf(sc[tn][j] - mi[j]);
                rs[j] += p;
                Ps[wave * 1024 + (j + 8 * hi) * 64 + tn * 16 + ln] = f2bf(p);
            }
        #pragma unroll
        for (int j = 0; j < 8; ++j) {
            #pragma unroll
            for (int m = 1; m < 16; m <<= 1) rs[j] += __shfl_xor(rs[j], m, 32);
            li[j] = li[j] * al[j] + rs[j];
        }
        #pragma unroll
        for (int t = 0; t < 4; ++t)
            #pragma unroll
            for (int j = 0; j < 8; ++j) o[t][j] *= al[j];

        // P (C-layout) was written to LDS; read back as A-fragments
        v16bf ap0, ap1;
        #pragma unroll
        for (int e = 0; e < 16; ++e) {
            ap0[e] = Ps[wave * 1024 + ln * 64 + akidx(e, hi)];
            ap1[e] = Ps[wave * 1024 + ln * 64 + akidx(e, hi) + 32];
        }
        #pragma unroll
        for (int t = 0; t < 4; ++t) {
            v16bf bv0, bv1;
            #pragma unroll
            for (int e = 0; e < 16; ++e) {
                bv0[e] = Vs[cur][akidx(e, hi) * 64 + t * 16 + ln];
                bv1[e] = Vs[cur][(akidx(e, hi) + 32) * 64 + t * 16 + ln];
            }
            o[t] = __builtin_amdgcn_wmma_f32_16x16x32_bf16(
                false, ap0, false, bv0, (short)0, o[t], false, false);
            o[t] = __builtin_amdgcn_wmma_f32_16x16x32_bf16(
                false, ap1, false, bv1, (short)0, o[t], false, false);
        }

        wait_async();
        __syncthreads();
    }

    // normalize and store: row (s*B+b), cols h*64 + [0,64)
    #pragma unroll
    for (int t = 0; t < 4; ++t)
        #pragma unroll
        for (int j = 0; j < 8; ++j) {
            const int s = q0 + wave * 16 + j + 8 * hi;
            const float v = o[t][j] / li[j];
            Out[((size_t)s * BATCH + b) * DMODEL + h * 64 + t * 16 + ln] = f2bf(v);
        }
}

// ---------------------------------------------------------------------------
// Host-side launch sequence
// ---------------------------------------------------------------------------
extern "C" void kernel_launch(void* const* d_in, const int* in_sizes, int n_in,
                              void* d_out, int out_size, void* d_ws, size_t ws_size,
                              hipStream_t stream) {
    const float* x       = (const float*)d_in[0];
    const float* norm1_w = (const float*)d_in[1];
    const float* wq      = (const float*)d_in[2];
    const float* bq      = (const float*)d_in[3];
    const float* wk      = (const float*)d_in[4];
    const float* bk      = (const float*)d_in[5];
    const float* wv      = (const float*)d_in[6];
    const float* bv      = (const float*)d_in[7];
    const float* qn_w    = (const float*)d_in[8];
    const float* kn_w    = (const float*)d_in[9];
    const float* wo      = (const float*)d_in[10];
    const float* bo      = (const float*)d_in[11];
    const float* norm2_w = (const float*)d_in[12];
    const float* w1      = (const float*)d_in[13];
    const float* b1      = (const float*)d_in[14];
    const float* w2      = (const float*)d_in[15];
    const float* b2      = (const float*)d_in[16];
    float* out           = (float*)d_out;

    char* ws = (char*)d_ws;
    // workspace layout (bytes)
    __bf16* h_bf  = (__bf16*)(ws + 0);           //  8388608
    __bf16* wq_bf = (__bf16*)(ws + 8388608);     //  2097152
    __bf16* wk_bf = (__bf16*)(ws + 10485760);    //   524288
    __bf16* wv_bf = (__bf16*)(ws + 11010048);    //   524288
    __bf16* wo_bf = (__bf16*)(ws + 11534336);    //  2097152
    __bf16* w1_bf = (__bf16*)(ws + 13631488);    //  8388608
    __bf16* w2_bf = (__bf16*)(ws + 22020096);    //  8388608
    float*  q_raw = (float*)(ws + 30408704);     // 16777216
    float*  k_raw = (float*)(ws + 47185920);     //  4194304
    float*  v_raw = (float*)(ws + 51380224);     //  4194304
    __bf16* qb    = (__bf16*)(ws + 55574528);    //  8388608
    __bf16* kb    = (__bf16*)(ws + 63963136);    //  2097152
    __bf16* vb    = (__bf16*)(ws + 66060288);    //  2097152
    __bf16* attn  = (__bf16*)(ws + 68157440);    //  8388608
    float*  x2    = (float*)(ws + 76546048);     // 16777216
    __bf16* h2_bf = (__bf16*)(ws + 93323264);    //  8388608
    // act aliases q_raw..vb region (those are dead before the MLP GEMMs)
    __bf16* act   = (__bf16*)(ws + 30408704);    // 33554432 (ends at 63963136)
    // total required: 101711872 bytes (~97 MB)

    // 1) weights -> bf16
    f32_to_bf16_kernel<<<(1024 * 1024 + 255) / 256, 256, 0, stream>>>(wq, wq_bf, 1024 * 1024);
    f32_to_bf16_kernel<<<(1024 * 256 + 255) / 256, 256, 0, stream>>>(wk, wk_bf, 1024 * 256);
    f32_to_bf16_kernel<<<(1024 * 256 + 255) / 256, 256, 0, stream>>>(wv, wv_bf, 1024 * 256);
    f32_to_bf16_kernel<<<(1024 * 1024 + 255) / 256, 256, 0, stream>>>(wo, wo_bf, 1024 * 1024);
    f32_to_bf16_kernel<<<(1024 * 4096 + 255) / 256, 256, 0, stream>>>(w1, w1_bf, 1024 * 4096);
    f32_to_bf16_kernel<<<(4096 * 1024 + 255) / 256, 256, 0, stream>>>(w2, w2_bf, 4096 * 1024);

    // 2) pre-attention RMSNorm
    rmsnorm_bf16_kernel<<<NROWS, 256, 0, stream>>>(x, norm1_w, h_bf);

    // 3) QKV projections (WMMA)
    gemm_bf16_wmma<EPI_F32><<<dim3(NROWS / 128, DMODEL / 128), 256, 0, stream>>>(
        h_bf, wq_bf, bq, nullptr, q_raw, NROWS, DMODEL, DMODEL);
    gemm_bf16_wmma<EPI_F32><<<dim3(NROWS / 128, 256 / 128), 256, 0, stream>>>(
        h_bf, wk_bf, bk, nullptr, k_raw, NROWS, 256, DMODEL);
    gemm_bf16_wmma<EPI_F32><<<dim3(NROWS / 128, 256 / 128), 256, 0, stream>>>(
        h_bf, wv_bf, bv, nullptr, v_raw, NROWS, 256, DMODEL);

    // 4) per-head QK RMSNorm + RoPE + relayout
    qkv_post_kernel<<<(NROWS * NHEADS + 2 * NROWS * NKV) / 8, 256, 0, stream>>>(
        q_raw, k_raw, v_raw, qn_w, kn_w, qb, kb, vb);

    // 5) causal GQA flash attention (WMMA)
    attn_wmma_kernel<<<dim3(S_LEN / 64, BATCH * NHEADS), 128, 0, stream>>>(qb, kb, vb, attn);

    // 6) output projection + residual
    gemm_bf16_wmma<EPI_F32_RES><<<dim3(NROWS / 128, DMODEL / 128), 256, 0, stream>>>(
        attn, wo_bf, bo, x, x2, NROWS, DMODEL, DMODEL);

    // 7) MLP: RMSNorm, up+GELU, down+residual
    rmsnorm_bf16_kernel<<<NROWS, 256, 0, stream>>>(x2, norm2_w, h2_bf);
    gemm_bf16_wmma<EPI_BF16_GELU><<<dim3(NROWS / 128, DFF / 128), 256, 0, stream>>>(
        h2_bf, w1_bf, b1, nullptr, act, NROWS, DFF, DMODEL);
    gemm_bf16_wmma<EPI_F32_RES><<<dim3(NROWS / 128, DMODEL / 128), 256, 0, stream>>>(
        act, w2_bf, b2, x2, out, NROWS, DMODEL, DFF);
}
